// TTRFluxLayer_32813550141486
// MI455X (gfx1250) — matrix operand order repaired and checked by reading, not compiled, not run on hardware
//
#include <hip/hip_runtime.h>

// ---------------------------------------------------------------------------
// CDNA5 (gfx1250) chunked bidirectional linear attention with MLP feature map.
// All GEMMs via v_wmma_f32_16x16x32_f16 (f16 operands, f32 accumulation).
// Scan is split over 2 D-halves per (head,dir) for 96-way WGP parallelism.
// ---------------------------------------------------------------------------

typedef _Float16 h8   __attribute__((ext_vector_type(8)));
typedef _Float16 v16h __attribute__((ext_vector_type(16)));
typedef float    v8f  __attribute__((ext_vector_type(8)));

union U16 { v16h v; h8 h[2]; };

#define DEVINL __device__ __forceinline__

// Problem dims (fixed by the reference)
constexpr int    kBH = 24, kN = 4096, kD = 128, kF = 256, kC = 128, kNC = 32;
constexpr int    kDH = kD / 2;                     // 64: D-half per scan WG
constexpr size_t kTOK = (size_t)kBH * kN;          // 98304 tokens

// LDS leading dims (padded to dodge bank conflicts; keep 16B row alignment)
constexpr int LDH  = kF + 8;  // phi hidden   [128][264] f16
constexpr int LDW  = kF + 8;  // wT f32 state [64][264]  f32
constexpr int LDWH = kF + 8;  // wT f16 shadow[64][264]  f16
constexpr int LDK  = kC + 8;  // kbT          [256][136] f16
constexpr int LDV  = kC + 8;  // vbT          [64][136]  f16
constexpr int LDA  = kC + 8;  // attn         [128][136] f16

constexpr size_t PHI_LDS = (size_t)kC * LDH * sizeof(_Float16);      // ~66 KB
constexpr size_t WT_B    = (size_t)kDH * LDW  * sizeof(float);       // 67584
constexpr size_t WTH_B   = (size_t)kDH * LDWH * sizeof(_Float16);    // 33792
constexpr size_t KBT_B   = (size_t)kF  * LDK  * sizeof(_Float16);    // 69632
constexpr size_t VBT_B   = (size_t)kDH * LDV  * sizeof(_Float16);    // 17408
constexpr size_t AT_B    = (size_t)kC  * LDA  * sizeof(_Float16);    // 34816
constexpr size_t SCAN_LDS = WT_B + WTH_B + KBT_B + VBT_B + AT_B;     // ~218 KB < 320 KB/WGP

// Workspace layout (bytes)
constexpr size_t W1H_OFF = 0;                                   // F*D f16
constexpr size_t W2H_OFF = W1H_OFF + (size_t)kF * kD * 2;       // F*F f16
constexpr size_t VH_OFF  = W2H_OFF + (size_t)kF * kF * 2;       // TOK*D f16
constexpr size_t QP_OFF  = VH_OFF  + kTOK * kD * 2;             // TOK*F f16
constexpr size_t KP_OFF  = QP_OFF  + kTOK * kF * 2;             // TOK*F f16
constexpr size_t REV_OFF = KP_OFF  + kTOK * kF * 2;             // TOK*D f32

DEVINL v8f wmma16(v16h a, v16h b, v8f c) {
  // D(16x16,f32) = A(16x32,f16) * B(32x16,f16) + C
  return __builtin_amdgcn_wmma_f32_16x16x32_f16(false, a, false, b, (short)0, c,
                                                false, false);
}

// --- WMMA operand builders -------------------------------------------------
// A (16x32 f16), row-major source, ld in elements.
// ISA layout: lane l: m=l&15, hf=l>>4; halves j=0..7 -> K=hf*8+j,
//             j=8..15 -> K=16+hf*8+(j-8)  => two contiguous 16B chunks.
DEVINL v16h loadA_h(const _Float16* base, int ld) {
  const int l = threadIdx.x & 31, m = l & 15, hf = l >> 4;
  U16 u;
  u.h[0] = *(const h8*)(base + (size_t)m * ld + hf * 8);
  u.h[1] = *(const h8*)(base + (size_t)m * ld + 16 + hf * 8);
  return u.v;
}
DEVINL v16h loadA_f(const float* base, int ld) {   // f32 source, cvt to f16
  const int l = threadIdx.x & 31, m = l & 15, hf = l >> 4;
  const float* p0 = base + (size_t)m * ld + hf * 8;
  const float* p1 = p0 + 16;
  U16 u;
#pragma unroll
  for (int j = 0; j < 8; ++j) { u.v[j] = (_Float16)p0[j]; u.v[8 + j] = (_Float16)p1[j]; }
  return u.v;
}
// B (32x16 f16), source must be column-major: baseT[n][k], ld in elements.
// ISA layout: lane l: n=l&15, hf=l>>4; j=0..7 -> K=hf*8+j, j=8..15 -> K=16+hf*8+(j-8).
DEVINL v16h loadB_h(const _Float16* baseT, int ld) {
  const int l = threadIdx.x & 31, n = l & 15, kb = (l >> 4) * 8;
  const h8* p = (const h8*)(baseT + (size_t)n * ld + kb);
  U16 u; u.h[0] = p[0]; u.h[1] = p[2];
  return u.v;
}

// ---------------------------------------------------------------------------
__global__ __launch_bounds__(256) void cvt_f32_f16_kernel(
    const float* __restrict__ src, _Float16* __restrict__ dst, int n) {
  int i = blockIdx.x * blockDim.x + threadIdx.x;
  const int stride = gridDim.x * blockDim.x;
  for (; i < n; i += stride) dst[i] = (_Float16)src[i];
}

// phi: per-128-token tile, OUT = (silu(X*W1^T + b1)) * W2^T + b2, stored f16.
__global__ __launch_bounds__(256) void phi_kernel(
    const float* __restrict__ q, const float* __restrict__ k,
    const _Float16* __restrict__ w1h, const _Float16* __restrict__ w2h,
    const float* __restrict__ b1, const float* __restrict__ b2,
    _Float16* __restrict__ qp, _Float16* __restrict__ kp) {
  extern __shared__ char smem[];
  _Float16* hs = (_Float16*)smem;                     // [128][LDH]

  const int bid = blockIdx.x;
  const int tiles = (int)(kTOK / kC);                 // 768 per tensor
  const bool isK = bid >= tiles;
  const float* src = isK ? k : q;
  _Float16* dst = isK ? kp : qp;
  const size_t t0 = (size_t)(isK ? bid - tiles : bid) * kC;

  const int tid = threadIdx.x, wv = tid >> 5, l = tid & 31;
  const int m = l & 15, hf = l >> 4;

  // ---- GEMM1: K=128 (4 x K32), A from global f32 (cvt), B = W1 [F][D] -----
  v16h a1[4];
#pragma unroll
  for (int kc = 0; kc < 4; ++kc)
    a1[kc] = loadA_f(src + (t0 + 16 * wv) * kD + kc * 32, kD);

#pragma unroll
  for (int nt = 0; nt < 16; ++nt) {
    v8f acc = {};
#pragma unroll
    for (int kc = 0; kc < 4; ++kc)
      acc = wmma16(a1[kc], loadB_h(w1h + (size_t)(nt * 16) * kD + kc * 32, kD), acc);
    const int n = nt * 16 + m;
    const float bias = b1[n];
#pragma unroll
    for (int r = 0; r < 8; ++r) {
      float x = acc[r] + bias;
      x = x / (1.0f + __expf(-x));                     // SiLU
      hs[(size_t)(16 * wv + 8 * hf + r) * LDH + n] = (_Float16)x;
    }
  }
  __syncthreads();

  // ---- GEMM2: K=256 (8 x K32), A from LDS, B = W2 [F][F] ------------------
  v16h a2[8];
#pragma unroll
  for (int kc = 0; kc < 8; ++kc)
    a2[kc] = loadA_h(hs + (size_t)(16 * wv) * LDH + kc * 32, LDH);

#pragma unroll
  for (int nt = 0; nt < 16; ++nt) {
    v8f acc = {};
#pragma unroll
    for (int kc = 0; kc < 8; ++kc)
      acc = wmma16(a2[kc], loadB_h(w2h + (size_t)(nt * 16) * kF + kc * 32, kF), acc);
    const int n = nt * 16 + m;
    const float bias = b2[n];
#pragma unroll
    for (int r = 0; r < 8; ++r)
      dst[(t0 + 16 * wv + 8 * hf + r) * kF + n] = (_Float16)(acc[r] + bias);
  }
}

// Chunked causal scan, one D-half per block.
// 96 blocks: head = bid%24, dir = (bid/24)%2, dhalf = bid/48.
__global__ __launch_bounds__(256) void scan_kernel(
    const _Float16* __restrict__ qp, const _Float16* __restrict__ kp,
    const _Float16* __restrict__ vh, float* __restrict__ outF,
    float* __restrict__ outR) {
  extern __shared__ char smem[];
  float*    wT  = (float*)smem;                            // [DH][LDW]  f32 state (w^T half)
  _Float16* wTh = (_Float16*)(smem + WT_B);                // [DH][LDWH] f16 shadow of wT
  _Float16* kbT = (_Float16*)(smem + WT_B + WTH_B);        // [F][LDK]
  _Float16* vbT = (_Float16*)(smem + WT_B + WTH_B + KBT_B);         // [DH][LDV]
  _Float16* at  = (_Float16*)(smem + WT_B + WTH_B + KBT_B + VBT_B); // [C][LDA] masked attn

  const int h = blockIdx.x % kBH;
  const int dir = (blockIdx.x / kBH) & 1;                  // 0 fwd, 1 rev
  const int dhalf = blockIdx.x / (2 * kBH);                // 0 or 1
  const int d0 = dhalf * kDH;
  const _Float16* qph = qp + (size_t)h * kN * kF;
  const _Float16* kph = kp + (size_t)h * kN * kF;
  const _Float16* vhh = vh + (size_t)h * kN * kD;
  float* out = (dir ? outR : outF) + (size_t)h * kN * kD;

  const int tid = threadIdx.x, wv = tid >> 5, l = tid & 31;
  const int m = l & 15, hf = l >> 4;
  const int mstr = wv & 3;          // update-phase d-stripe (0..3)
  const int nt0  = (wv >> 2) * 8;   // update-phase nt range start (0 or 8)

  for (int i = tid; i < kDH * LDW; i += 256) wT[i] = 0.0f; // zero f32 state
  for (int i = tid; i < kDH * LDWH; i += 256) wTh[i] = (_Float16)0.0f;
  __syncthreads();

  for (int step = 0; step < kNC; ++step) {
    const int ci = dir ? (kNC - 1 - step) : step;          // rev walks backwards
    const int t0 = ci * kC;

    // stage vb^T [d][c] (this half) and kb^T [f][c] into LDS
    for (int idx = tid; idx < kC * kDH; idx += 256) {
      const int c = idx >> 6, d = idx & 63;
      vbT[(size_t)d * LDV + c] = vhh[(size_t)(t0 + c) * kD + d0 + d];
    }
    for (int idx = tid; idx < kC * kF; idx += 256) {
      const int c = idx >> 8, f = idx & 255;
      kbT[(size_t)f * LDK + c] = kph[(size_t)(t0 + c) * kF + f];
    }
    // prefetch next chunk's streams into cache
    if (step + 1 < kNC && l == 0) {
      const int tn = (dir ? (ci - 1) : (ci + 1)) * kC;
      __builtin_prefetch(qph + (size_t)(tn + 16 * wv) * kF, 0, 1);
      __builtin_prefetch(kph + (size_t)(tn + 16 * wv) * kF, 0, 1);
      __builtin_prefetch(vhh + (size_t)(tn + 16 * wv) * kD, 0, 1);
    }
    __syncthreads();

    // A operands: this wave's 16-row stripe of qb, full K=F (8 x K32)
    v16h aq[8];
#pragma unroll
    for (int kc = 0; kc < 8; ++kc)
      aq[kc] = loadA_h(qph + (size_t)(t0 + 16 * wv) * kF + kc * 32, kF);

    // attn = qb * kb^T, masked (fwd: s<=c, rev: s>=c), f16 into LDS
#pragma unroll
    for (int st = 0; st < 8; ++st) {
      v8f acc = {};
#pragma unroll
      for (int kc = 0; kc < 8; ++kc)
        acc = wmma16(aq[kc], loadB_h(kph + (size_t)(t0 + st * 16) * kF + kc * 32, kF), acc);
#pragma unroll
      for (int r = 0; r < 8; ++r) {
        const int cl = 16 * wv + 8 * hf + r;
        const int sl = st * 16 + m;
        const bool keep = dir ? (sl >= cl) : (sl <= cl);
        at[(size_t)cl * LDA + sl] = (_Float16)(keep ? acc[r] : 0.0f);
      }
    }

    // inter = qb * w-half  (B = f16 shadow of w^T, straight LDS b128 loads)
    v8f io[4];
#pragma unroll
    for (int dt = 0; dt < 4; ++dt) {
      v8f acc = {};
#pragma unroll
      for (int kc = 0; kc < 8; ++kc)
        acc = wmma16(aq[kc], loadB_h(wTh + (size_t)(dt * 16) * LDWH + kc * 32, LDWH), acc);
      io[dt] = acc;
    }
    __syncthreads();   // attn visible to all; all wTh reads for this chunk done

    // intra: io += attn * vb-half   (A = attn rows, B = vb^T [d][s])
    v16h aa[4];
#pragma unroll
    for (int kc = 0; kc < 4; ++kc)
      aa[kc] = loadA_h(at + (size_t)(16 * wv) * LDA + kc * 32, LDA);
#pragma unroll
    for (int dt = 0; dt < 4; ++dt) {
#pragma unroll
      for (int kc = 0; kc < 4; ++kc)
        io[dt] = wmma16(aa[kc], loadB_h(vbT + (size_t)(dt * 16) * LDV + kc * 32, LDV), io[dt]);
    }

    // out = (inter + intra) / counts   (only this D-half's columns)
#pragma unroll
    for (int dt = 0; dt < 4; ++dt) {
#pragma unroll
      for (int r = 0; r < 8; ++r) {
        const int cl = 16 * wv + 8 * hf + r;
        const int t = t0 + cl;
        const int d = d0 + dt * 16 + m;
        const float cnt = dir ? (float)(kN - t) : (float)(t + 1);
        out[(size_t)t * kD + d] = io[dt][r] / cnt;
      }
    }

    // state update: wT[d][f] += sum_c vbT[d][c] * kbT[f][c]; refresh f16 shadow.
    // Wave wv owns d-stripe (wv&3) and nt range [(wv>>2)*8, +8): unique owners.
    v16h av[4];
#pragma unroll
    for (int kc = 0; kc < 4; ++kc)
      av[kc] = loadA_h(vbT + (size_t)(16 * mstr) * LDV + kc * 32, LDV);
#pragma unroll
    for (int nti = 0; nti < 8; ++nti) {
      const int nt = nt0 + nti;
      v8f u = {};
#pragma unroll
      for (int kc = 0; kc < 4; ++kc)
        u = wmma16(av[kc], loadB_h(kbT + (size_t)(nt * 16) * LDK + kc * 32, LDK), u);
#pragma unroll
      for (int r = 0; r < 8; ++r) {
        const int d = 16 * mstr + 8 * hf + r;
        const int f = nt * 16 + m;
        const float nv = wT[(size_t)d * LDW + f] + u[r];
        wT[(size_t)d * LDW + f] = nv;          // f32 master
        wTh[(size_t)d * LDWH + f] = (_Float16)nv;  // f16 shadow for WMMA feed
      }
    }
    __syncthreads();   // update done before next chunk's staging/inter
  }
}

__global__ __launch_bounds__(256) void add_kernel(
    float* __restrict__ out, const float* __restrict__ rev, int n) {
  int i = blockIdx.x * blockDim.x + threadIdx.x;
  const int stride = gridDim.x * blockDim.x;
  for (; i < n; i += stride) out[i] += rev[i];
}

// ---------------------------------------------------------------------------
extern "C" void kernel_launch(void* const* d_in, const int* in_sizes, int n_in,
                              void* d_out, int out_size, void* d_ws, size_t ws_size,
                              hipStream_t stream) {
  const float* q  = (const float*)d_in[0];
  const float* k  = (const float*)d_in[1];
  const float* v  = (const float*)d_in[2];
  const float* w1 = (const float*)d_in[3];
  const float* b1 = (const float*)d_in[4];
  const float* w2 = (const float*)d_in[5];
  const float* b2 = (const float*)d_in[6];

  char* ws = (char*)d_ws;
  _Float16* w1h = (_Float16*)(ws + W1H_OFF);
  _Float16* w2h = (_Float16*)(ws + W2H_OFF);
  _Float16* vh  = (_Float16*)(ws + VH_OFF);
  _Float16* qpp = (_Float16*)(ws + QP_OFF);
  _Float16* kpp = (_Float16*)(ws + KP_OFF);
  float*    rev = (float*)(ws + REV_OFF);
  float*    out = (float*)d_out;

  (void)hipFuncSetAttribute((const void*)phi_kernel,
                            hipFuncAttributeMaxDynamicSharedMemorySize, (int)PHI_LDS);
  (void)hipFuncSetAttribute((const void*)scan_kernel,
                            hipFuncAttributeMaxDynamicSharedMemorySize, (int)SCAN_LDS);

  // 1) f32 -> f16 conversions (weights + v)
  cvt_f32_f16_kernel<<<(kF * kD + 255) / 256, 256, 0, stream>>>(w1, w1h, kF * kD);
  cvt_f32_f16_kernel<<<(kF * kF + 255) / 256, 256, 0, stream>>>(w2, w2h, kF * kF);
  cvt_f32_f16_kernel<<<(int)((kTOK * kD + 255) / 256), 256, 0, stream>>>(v, vh, (int)(kTOK * kD));

  // 2) feature map phi on q and k (768 tiles each)
  phi_kernel<<<2 * (int)(kTOK / kC), 256, PHI_LDS, stream>>>(q, k, w1h, w2h, b1, b2, qpp, kpp);

  // 3) bidirectional chunked scan (24 heads x 2 directions x 2 D-halves)
  scan_kernel<<<4 * kBH, 256, SCAN_LDS, stream>>>(qpp, kpp, vh, out, rev);

  // 4) merge forward + reverse
  add_kernel<<<(int)((kTOK * kD + 255) / 256), 256, 0, stream>>>(out, rev, (int)(kTOK * kD));
}
